// EGC_Stage2_Single_3676492005551
// MI455X (gfx1250) — matrix operand order repaired and checked by reading, not compiled, not run on hardware
//
#include <hip/hip_runtime.h>
#include <hip/hip_bf16.h>

#define B_    8
#define T_    12
#define N_    200
#define C_IN  2
#define H_    64
#define D_    32
#define NP    224      // padded node dim (multiple of 32 for WMMA K)
#define CP    96       // padded per-hop channel dim (multiple of 32)
#define GP    (3*CP)   // 288: concatenated graph-conv feature dim (padded)
#define RU_O  128
#define CAND_O 64

typedef _Float16 h16 __attribute__((ext_vector_type(16)));
typedef _Float16 h8  __attribute__((ext_vector_type(8)));
typedef float    f8  __attribute__((ext_vector_type(8)));

union FragH { h16 v; h8 h[2]; };

// ---- WMMA fragment loaders (v_wmma_f32_16x16x32_f16 layouts) ----
// A fragment: lane m = row0 + (lane&15); halves 0..7 -> K = hs*8+e,
// halves 8..15 -> K = 16+hs*8+(e-8). Source row-major (row, K), ld in halfs.
__device__ inline h16 load_a_frag(const _Float16* base, int row0, int kBase, int ld) {
  int lane = threadIdx.x & 31;
  const _Float16* p = base + (size_t)(row0 + (lane & 15)) * ld + kBase + ((lane >> 4) * 8);
  FragH f;
  f.h[0] = *(const h8*)(p);
  f.h[1] = *(const h8*)(p + 16);
  return f.v;
}
// B fragment: lane n = col0 + (lane&15); halves 0..7 -> K = hs*8+e,
// halves 8..15 -> K = 16+hs*8+(e-8). Source stored "B-transposed": (n, K), ld in halfs.
__device__ inline h16 load_b_frag(const _Float16* baseT, int col0, int kBase, int ld) {
  int lane = threadIdx.x & 31;
  const _Float16* p = baseT + (size_t)(col0 + (lane & 15)) * ld + kBase + ((lane >> 4) * 8);
  FragH f;
  f.h[0] = *(const h8*)(p);
  f.h[1] = *(const h8*)(p + 16);
  return f.v;
}

__device__ inline float sigmoidf_(float x) { return 1.f / (1.f + __expf(-x)); }

// ---------------- init kernels ----------------
__global__ void k_init_state(float* state) {
  int i = blockIdx.x * blockDim.x + threadIdx.x;
  if (i < B_ * N_ * H_) state[i] = 0.f;
}

__global__ void k_init_dy(const float* __restrict__ nf, const float* __restrict__ W,
                          const float* __restrict__ b, float* __restrict__ dy) {
  int i = blockIdx.x * blockDim.x + threadIdx.x;
  if (i >= B_ * N_ * D_) return;
  int d = i % D_;
  int n = (i / D_) % N_;
  float a = b[d];
  for (int k = 0; k < 64; ++k) a += nf[n * 64 + k] * W[k * D_ + d];
  dy[i] = a;
}

// Build f16 B-transposed weights for the feature GEMM, with zero channel padding.
__global__ void k_prep_weights(const float* __restrict__ Wru, const float* __restrict__ Wcand,
                               _Float16* __restrict__ WruT, _Float16* __restrict__ WcandT) {
  int i = blockIdx.x * blockDim.x + threadIdx.x;
  const int t1 = RU_O * GP;
  const int t2 = CAND_O * GP;
  if (i < t1) {
    int o = i / GP, kp = i % GP, hop = kp / CP, c = kp % CP;
    float v = (c < C_IN + H_) ? Wru[(hop * (C_IN + H_) + c) * RU_O + o] : 0.f;
    WruT[i] = (_Float16)v;
  } else if (i < t1 + t2) {
    int j = i - t1;
    int o = j / GP, kp = j % GP, hop = kp / CP, c = kp % CP;
    float v = (c < C_IN + H_) ? Wcand[(hop * (C_IN + H_) + c) * CAND_O + o] : 0.f;
    WcandT[j] = (_Float16)v;
  }
}

// ---------------- per-step dynamic-state GRU + pair-MLP projections ----------------
__global__ void k_gru_dy(const float* __restrict__ x, int t, float* __restrict__ state_dy,
                         const float* __restrict__ W_rz, const float* __restrict__ b_rz,
                         const float* __restrict__ W_h, const float* __restrict__ b_h,
                         const float* __restrict__ Wc2, const float* __restrict__ Wm2,
                         float* __restrict__ Pgi, float* __restrict__ Pgj,
                         float* __restrict__ Pmi, float* __restrict__ Pmj) {
  int idx = blockIdx.x * blockDim.x + threadIdx.x;
  if (idx >= B_ * N_) return;
  int n = idx % N_, b = idx / N_;
  float inp = x[(((size_t)b * T_ + t) * N_ + n) * C_IN + 0];
  float dy[D_];
  for (int d = 0; d < D_; ++d) dy[d] = state_dy[(size_t)idx * D_ + d];
  float r[D_], z[D_];
  for (int j = 0; j < 2 * D_; ++j) {
    float a = inp * W_rz[j] + b_rz[j];
    for (int d = 0; d < D_; ++d) a += dy[d] * W_rz[(1 + d) * (2 * D_) + j];
    float sg = sigmoidf_(a);
    if (j < D_) r[j] = sg; else z[j - D_] = sg;
  }
  float rdy[D_];
  for (int d = 0; d < D_; ++d) rdy[d] = r[d] * dy[d];
  float s[D_];
  for (int j = 0; j < D_; ++j) {
    float a = inp * W_h[j] + b_h[j];
    for (int d = 0; d < D_; ++d) a += rdy[d] * W_h[(1 + d) * D_ + j];
    float hh = tanhf(a);
    float nd = z[j] * dy[j] + (1.f - z[j]) * hh;
    state_dy[(size_t)idx * D_ + j] = nd;
    s[j] = fmaxf(nd, 0.f);
  }
  for (int c = 0; c < D_; ++c) {
    float g1 = 0.f, g2 = 0.f, m1 = 0.f, m2 = 0.f;
    for (int d = 0; d < D_; ++d) {
      g1 += s[d] * Wc2[d * D_ + c];
      g2 += s[d] * Wc2[(D_ + d) * D_ + c];
      m1 += s[d] * Wm2[d * D_ + c];
      m2 += s[d] * Wm2[(D_ + d) * D_ + c];
    }
    Pgi[(size_t)idx * D_ + c] = g1;
    Pgj[(size_t)idx * D_ + c] = g2;
    Pmi[(size_t)idx * D_ + c] = m1;
    Pmj[(size_t)idx * D_ + c] = m2;
  }
}

// ---------------- pairwise adjacency: A = g * sigmoid(m) ----------------
__global__ void k_pair_A(const float* __restrict__ Pgi, const float* __restrict__ Pgj,
                         const float* __restrict__ Pmi, const float* __restrict__ Pmj,
                         const float* __restrict__ bc2, const float* __restrict__ Wc1,
                         const float* __restrict__ bc1, const float* __restrict__ bm2,
                         const float* __restrict__ Wm1, const float* __restrict__ bm1,
                         float* __restrict__ outA, _Float16* __restrict__ Ah) {
  __shared__ float sPgi[16][D_], sPgj[16][D_], sPmi[16][D_], sPmj[16][D_];
  __shared__ float sbc2[D_], sWc1[D_], sbm2[D_], sWm1[D_];
  int b = blockIdx.x, it = blockIdx.y, jt = blockIdx.z;
  int tid = threadIdx.x;
  for (int sdx = tid; sdx < 16 * D_; sdx += 256) {
    int r = sdx >> 5, c = sdx & 31;
    int gi = it * 16 + r, gj = jt * 16 + r;
    sPgi[r][c] = (gi < N_) ? Pgi[((size_t)b * N_ + gi) * D_ + c] : 0.f;
    sPmi[r][c] = (gi < N_) ? Pmi[((size_t)b * N_ + gi) * D_ + c] : 0.f;
    sPgj[r][c] = (gj < N_) ? Pgj[((size_t)b * N_ + gj) * D_ + c] : 0.f;
    sPmj[r][c] = (gj < N_) ? Pmj[((size_t)b * N_ + gj) * D_ + c] : 0.f;
  }
  if (tid < D_) { sbc2[tid] = bc2[tid]; sWc1[tid] = Wc1[tid]; sbm2[tid] = bm2[tid]; sWm1[tid] = Wm1[tid]; }
  __syncthreads();
  int li = tid >> 4, lj = tid & 15;
  int gi = it * 16 + li, gj = jt * 16 + lj;
  float Aval = 0.f;
  if (gi < N_ && gj < N_) {
    float g = bc1[0], m = bm1[0];
#pragma unroll
    for (int c = 0; c < D_; ++c) {
      g += fmaxf(sPgi[li][c] + sPgj[lj][c] + sbc2[c], 0.f) * sWc1[c];
      m += fmaxf(sPmi[li][c] + sPmj[lj][c] + sbm2[c], 0.f) * sWm1[c];
    }
    Aval = g * sigmoidf_(m);
    outA[((size_t)b * N_ + gi) * N_ + gj] = Aval;
  }
  Ah[((size_t)b * NP + gi) * NP + gj] = (_Float16)Aval;
}

// ---------------- pack graph-conv input (both layouts, f16, zero-padded) ----------------
__global__ void k_pack_x0(const float* __restrict__ x, const float* __restrict__ state,
                          const float* __restrict__ RuT, int mode, int t,
                          _Float16* __restrict__ X0t, _Float16* __restrict__ X0n) {
  int b = blockIdx.x, i = blockIdx.y;
  int c = threadIdx.x;
  if (c >= CP) return;
  float val = 0.f;
  if (i < N_) {
    if (c < C_IN) {
      val = x[(((size_t)b * T_ + t) * N_ + i) * C_IN + c];
    } else if (c < C_IN + H_) {
      int h = c - C_IN;
      float st = state[((size_t)b * N_ + i) * H_ + h];
      if (mode == 0) val = st;
      else val = RuT[((size_t)b * RU_O + h) * NP + i] * st;  // rr * state
    }
  }
  _Float16 hv = (_Float16)val;
  X0t[((size_t)b * CP + c) * NP + i] = hv;  // channel-major (c, i)
  X0n[((size_t)b * NP + i) * CP + c] = hv;  // node-major    (i, c)
}

// ---------------- WMMA hop: Y = A @ X  (M=i, K=j, N=c) ----------------
__global__ void k_hop_gemm(const _Float16* __restrict__ Ah,   // (B, NP, NP) row-major (i,j)
                           const _Float16* __restrict__ Xt,   // (B, CP, NP) (c, j)
                           _Float16* __restrict__ Yt,         // (B, CP, NP) (c, i)
                           _Float16* __restrict__ Yn) {       // (B, NP, CP) (i, c)
  int b = blockIdx.x;
  int mt = blockIdx.y;
  int wave = threadIdx.x >> 5;
  if (wave >= CP / 16) return;
  int nt = wave;
  const _Float16* A = Ah + (size_t)b * NP * NP;
  const _Float16* X = Xt + (size_t)b * CP * NP;
  f8 acc = {};
  for (int k = 0; k < NP; k += 32) {
    h16 fa = load_a_frag(A, mt * 16, k, NP);
    h16 fb = load_b_frag(X, nt * 16, k, NP);
    acc = __builtin_amdgcn_wmma_f32_16x16x32_f16(false, fa, false, fb, (short)0, acc, false, false);
  }
  int lane = threadIdx.x & 31;
  int c  = nt * 16 + (lane & 15);
  int i0 = mt * 16 + ((lane >> 4) * 8);
  h8 hv;
#pragma unroll
  for (int v = 0; v < 8; ++v) hv[v] = (_Float16)acc[v];
  *(h8*)(Yt + (size_t)b * CP * NP + (size_t)c * NP + i0) = hv;  // contiguous in i
  _Float16* yn = Yn + (size_t)b * NP * CP;
#pragma unroll
  for (int v = 0; v < 8; ++v) yn[(size_t)(i0 + v) * CP + c] = hv[v];
}

// ---------------- WMMA feature GEMM: Out = act(feats @ W + b), transposed output ----------------
__global__ void k_feat_gemm(const _Float16* __restrict__ X0n, const _Float16* __restrict__ X1n,
                            const _Float16* __restrict__ X2n, const _Float16* __restrict__ Wt,
                            const float* __restrict__ bias, float* __restrict__ OutT,
                            int OUT, int act) {  // act: 0=sigmoid, 1=tanh
  int b = blockIdx.x;
  int mt = blockIdx.y;
  int wave = threadIdx.x >> 5;
  if (wave >= OUT / 16) return;
  int nt = wave;
  const _Float16* Xs[3] = { X0n + (size_t)b * NP * CP,
                            X1n + (size_t)b * NP * CP,
                            X2n + (size_t)b * NP * CP };
  f8 acc = {};
  for (int kc = 0; kc < GP / 32; ++kc) {
    int hop = kc / (CP / 32);
    int kin = (kc % (CP / 32)) * 32;
    h16 fa = load_a_frag(Xs[hop], mt * 16, kin, CP);
    h16 fb = load_b_frag(Wt, nt * 16, kc * 32, GP);
    acc = __builtin_amdgcn_wmma_f32_16x16x32_f16(false, fa, false, fb, (short)0, acc, false, false);
  }
  int lane = threadIdx.x & 31;
  int o  = nt * 16 + (lane & 15);
  int i0 = mt * 16 + ((lane >> 4) * 8);
  float bo = bias[o];
  f8 r;
#pragma unroll
  for (int v = 0; v < 8; ++v) {
    float xv = acc[v] + bo;
    r[v] = act ? tanhf(xv) : sigmoidf_(xv);
  }
  *(f8*)(OutT + (size_t)b * OUT * NP + (size_t)o * NP + i0) = r;
}

// ---------------- GRU state update ----------------
__global__ void k_state_update(float* __restrict__ state, const float* __restrict__ RuT,
                               const float* __restrict__ CT) {
  int idx = blockIdx.x * blockDim.x + threadIdx.x;
  if (idx >= B_ * N_ * H_) return;
  int h = idx % H_;
  int n = (idx / H_) % N_;
  int b = idx / (H_ * N_);
  float u = RuT[((size_t)b * RU_O + (H_ + h)) * NP + n];
  float c = CT[((size_t)b * CAND_O + h) * NP + n];
  state[idx] = u * state[idx] + (1.f - u) * c;
}

// ---------------- final prediction head ----------------
__global__ void k_pred(const float* __restrict__ state, const float* __restrict__ Wp1,
                       const float* __restrict__ bp1, const float* __restrict__ Wp2,
                       const float* __restrict__ bp2, float* __restrict__ out) {
  int idx = blockIdx.x * blockDim.x + threadIdx.x;
  if (idx >= B_ * N_) return;
  float hv[H_];
  for (int k = 0; k < H_; ++k) hv[k] = state[(size_t)idx * H_ + k];
  float acc = bp2[0];
  for (int j = 0; j < H_; ++j) {
    float a = bp1[j];
    for (int k = 0; k < H_; ++k) a += hv[k] * Wp1[k * H_ + j];
    a = (a > 0.f) ? a : 0.01f * a;  // leaky_relu, slope 0.01
    acc += a * Wp2[j];
  }
  out[idx] = acc;
}

extern "C" void kernel_launch(void* const* d_in, const int* in_sizes, int n_in,
                              void* d_out, int out_size, void* d_ws, size_t ws_size,
                              hipStream_t stream) {
  const float* x           = (const float*)d_in[0];
  const float* node_fea_st = (const float*)d_in[1];
  const float* W_s2d = (const float*)d_in[2];  const float* b_s2d = (const float*)d_in[3];
  const float* W_rz  = (const float*)d_in[4];  const float* b_rz  = (const float*)d_in[5];
  const float* W_h   = (const float*)d_in[6];  const float* b_h   = (const float*)d_in[7];
  const float* Wc2   = (const float*)d_in[8];  const float* bc2   = (const float*)d_in[9];
  const float* Wc1   = (const float*)d_in[10]; const float* bc1   = (const float*)d_in[11];
  const float* Wm2   = (const float*)d_in[12]; const float* bm2   = (const float*)d_in[13];
  const float* Wm1   = (const float*)d_in[14]; const float* bm1   = (const float*)d_in[15];
  const float* W_ru  = (const float*)d_in[16]; const float* b_ru  = (const float*)d_in[17];
  const float* W_cand= (const float*)d_in[18]; const float* b_cand= (const float*)d_in[19];
  const float* Wp1   = (const float*)d_in[20]; const float* bp1   = (const float*)d_in[21];
  const float* Wp2   = (const float*)d_in[22]; const float* bp2   = (const float*)d_in[23];

  float* predOut = (float*)d_out;            // (B, N, 1)
  float* dyG     = (float*)d_out + B_ * N_;  // (T, B, N, N)

  char* p = (char*)d_ws;
  auto alloc = [&](size_t bytes) { void* r = (void*)p; p += (bytes + 255) & ~(size_t)255; return r; };
  float*    state_dy = (float*)alloc((size_t)B_ * N_ * D_ * 4);
  float*    state    = (float*)alloc((size_t)B_ * N_ * H_ * 4);
  float*    Pgi = (float*)alloc((size_t)B_ * N_ * D_ * 4);
  float*    Pgj = (float*)alloc((size_t)B_ * N_ * D_ * 4);
  float*    Pmi = (float*)alloc((size_t)B_ * N_ * D_ * 4);
  float*    Pmj = (float*)alloc((size_t)B_ * N_ * D_ * 4);
  float*    RuT = (float*)alloc((size_t)B_ * RU_O * NP * 4);
  float*    CT  = (float*)alloc((size_t)B_ * CAND_O * NP * 4);
  _Float16* Ah  = (_Float16*)alloc((size_t)B_ * NP * NP * 2);
  _Float16* X0t = (_Float16*)alloc((size_t)B_ * CP * NP * 2);
  _Float16* X1t = (_Float16*)alloc((size_t)B_ * CP * NP * 2);
  _Float16* X2t = (_Float16*)alloc((size_t)B_ * CP * NP * 2);
  _Float16* X0n = (_Float16*)alloc((size_t)B_ * NP * CP * 2);
  _Float16* X1n = (_Float16*)alloc((size_t)B_ * NP * CP * 2);
  _Float16* X2n = (_Float16*)alloc((size_t)B_ * NP * CP * 2);
  _Float16* WruT   = (_Float16*)alloc((size_t)RU_O * GP * 2);
  _Float16* WcandT = (_Float16*)alloc((size_t)CAND_O * GP * 2);

  k_prep_weights<<<dim3(((RU_O + CAND_O) * GP + 255) / 256), 256, 0, stream>>>(W_ru, W_cand, WruT, WcandT);
  k_init_state<<<(B_ * N_ * H_ + 255) / 256, 256, 0, stream>>>(state);
  k_init_dy<<<(B_ * N_ * D_ + 255) / 256, 256, 0, stream>>>(node_fea_st, W_s2d, b_s2d, state_dy);

  for (int t = 0; t < T_; ++t) {
    k_gru_dy<<<(B_ * N_ + 127) / 128, 128, 0, stream>>>(x, t, state_dy, W_rz, b_rz, W_h, b_h,
                                                        Wc2, Wm2, Pgi, Pgj, Pmi, Pmj);
    k_pair_A<<<dim3(B_, NP / 16, NP / 16), 256, 0, stream>>>(Pgi, Pgj, Pmi, Pmj,
                                                             bc2, Wc1, bc1, bm2, Wm1, bm1,
                                                             dyG + (size_t)t * B_ * N_ * N_, Ah);
    // ---- ru pass: graph_conv(concat(xt, state)) ----
    k_pack_x0<<<dim3(B_, NP), 128, 0, stream>>>(x, state, RuT, 0, t, X0t, X0n);
    k_hop_gemm<<<dim3(B_, NP / 16), 192, 0, stream>>>(Ah, X0t, X1t, X1n);
    k_hop_gemm<<<dim3(B_, NP / 16), 192, 0, stream>>>(Ah, X1t, X2t, X2n);
    k_feat_gemm<<<dim3(B_, NP / 16), 256, 0, stream>>>(X0n, X1n, X2n, WruT, b_ru, RuT, RU_O, 0);
    // ---- candidate pass: graph_conv(concat(xt, rr*state)) ----
    k_pack_x0<<<dim3(B_, NP), 128, 0, stream>>>(x, state, RuT, 1, t, X0t, X0n);
    k_hop_gemm<<<dim3(B_, NP / 16), 192, 0, stream>>>(Ah, X0t, X1t, X1n);
    k_hop_gemm<<<dim3(B_, NP / 16), 192, 0, stream>>>(Ah, X1t, X2t, X2n);
    k_feat_gemm<<<dim3(B_, NP / 16), 128, 0, stream>>>(X0n, X1n, X2n, WcandT, b_cand, CT, CAND_O, 1);
    k_state_update<<<(B_ * N_ * H_ + 255) / 256, 256, 0, stream>>>(state, RuT, CT);
  }

  k_pred<<<(B_ * N_ + 127) / 128, 128, 0, stream>>>(state, Wp1, bp1, Wp2, bp2, predOut);
}
